// TemporalAttentionLayer_61684320305211
// MI455X (gfx1250) — matrix operand order repaired and checked by reading, not compiled, not run on hardware
//
#include <hip/hip_runtime.h>

#define N_NODES 100000
#define N_EDGES 600000
#define HIDDEN  128
#define NTILES  (N_EDGES / 16)   // 37500, exact

typedef __attribute__((ext_vector_type(16))) __bf16 v16bf;
typedef __attribute__((ext_vector_type(8)))  float  v8f;
typedef __attribute__((ext_vector_type(4)))  int    v4i;

union BfPack { uint4 q[2]; v16bf v; };

__device__ __forceinline__ unsigned short f2bf(float f) {
    unsigned u = __float_as_uint(f);
    unsigned r = (u + 0x7FFFu + ((u >> 16) & 1u)) >> 16;
    return (unsigned short)r;
}

// ---------------------------------------------------------------------------
// Kernel 1: pack W1[0:256][:] (row-major f32) into bf16 B-operand tiles.
// Tile t = ((half*4 + kstep)*8 + ntile), 512 bf16 per tile.
// Within tile: lane l (0..31) holds 16 contiguous bf16 = column n = nt*16+(l&15),
// K rows (l<16 ? 0..7,16..23 : 8..15,24..31) of that 32-deep k-slab.
// ---------------------------------------------------------------------------
__global__ void pack_w1_kernel(const float* __restrict__ W1,
                               unsigned short* __restrict__ W1B) {
    for (int idx = threadIdx.x; idx < 2 * 4 * 8 * 512; idx += blockDim.x) {
        int t      = idx >> 9;
        int within = idx & 511;
        int l = within >> 4;
        int i = within & 15;
        int h  = t >> 5;
        int ks = (t >> 3) & 3;
        int nt = t & 7;
        int n    = nt * 16 + (l & 15);
        int kloc = ((l >> 4) << 3) + (i & 7) + ((i >> 3) << 4);  // 0..7,16..23 or 8..15,24..31
        int krow = h * HIDDEN + ks * 32 + kloc;  // 0..255
        W1B[idx] = f2bf(W1[krow * HIDDEN + n]);
    }
}

// ---------------------------------------------------------------------------
// Kernel 2: x -> bf16 copy (A-operand source), out = x, sum_exp = 0
// ---------------------------------------------------------------------------
__global__ void convert_x_kernel(const float* __restrict__ x,
                                 unsigned short* __restrict__ xq,
                                 float* __restrict__ out,
                                 float* __restrict__ sum_exp) {
    long long i = (long long)blockIdx.x * blockDim.x + threadIdx.x;
    long long base = i * 4;
    if (base < (long long)N_NODES * HIDDEN) {
        float4 v = *(const float4*)(x + base);
        *(float4*)(out + base) = v;
        unsigned lo = (unsigned)f2bf(v.x) | ((unsigned)f2bf(v.y) << 16);
        unsigned hi = (unsigned)f2bf(v.z) | ((unsigned)f2bf(v.w) << 16);
        uint2 p; p.x = lo; p.y = hi;
        *(uint2*)(xq + base) = p;
    }
    if (i < N_NODES) sum_exp[i] = 0.0f;
}

// ---------------------------------------------------------------------------
// Kernel 3: per-edge MLP scores via bf16 WMMA. One wave per 16-edge tile.
// ---------------------------------------------------------------------------
__device__ __forceinline__ v16bf load_a(const unsigned short* row, int kk, int lane) {
    int base = kk + ((lane >> 4) << 3);  // +8 shorts for upper half-lanes
    BfPack u;
    u.q[0] = *(const uint4*)(row + base);       // K base+0..7   -> VGPR 0..3
    u.q[1] = *(const uint4*)(row + base + 16);  // K base+16..23 -> VGPR 4..7
    return u.v;
}

__device__ __forceinline__ v16bf load_b(const unsigned short* ldsW,
                                        int half, int ks, int nt, int lane) {
    int t = (half * 4 + ks) * 8 + nt;
    const uint4* p = (const uint4*)(ldsW + t * 512 + lane * 16);
    BfPack u;
    u.q[0] = p[0];
    u.q[1] = p[1];
    return u.v;
}

__global__ void __launch_bounds__(256)
edge_score_kernel(const unsigned short* __restrict__ xq,
                  const unsigned short* __restrict__ W1B,
                  const long long* __restrict__ edge_index,
                  const float* __restrict__ edge_weight,
                  const float* __restrict__ b1,
                  const float* __restrict__ w1c,   // W1 row 256
                  const float* __restrict__ W2,
                  const float* __restrict__ b2,
                  float* __restrict__ exp_scores,
                  float* __restrict__ sum_exp) {
    __shared__ unsigned short ldsW[2 * 4 * 8 * 512];  // 64 KB

    // Stage packed W1 into LDS once per workgroup via the gfx1250 direct
    // async global->LDS path (no VGPR round-trip, ASYNCcnt-tracked).
#if __has_builtin(__builtin_amdgcn_global_load_async_to_lds_b128)
    {
        typedef __attribute__((address_space(1))) v4i g_v4i;
        typedef __attribute__((address_space(3))) v4i l_v4i;
        g_v4i* gp = (g_v4i*)W1B;   // 16-byte chunks in global
        l_v4i* lp = (l_v4i*)ldsW;  // 16-byte chunks in LDS
        for (int ci = threadIdx.x; ci < (2 * 4 * 8 * 512 * 2) / 16; ci += blockDim.x)
            __builtin_amdgcn_global_load_async_to_lds_b128(gp + ci, lp + ci, 0, 0);
#if __has_builtin(__builtin_amdgcn_s_wait_asynccnt)
        __builtin_amdgcn_s_wait_asynccnt(0);
#else
        asm volatile("s_wait_asynccnt 0" ::: "memory");
#endif
    }
#else
    {
        const uint4* g = (const uint4*)W1B;
        uint4* s = (uint4*)ldsW;
        for (int i = threadIdx.x; i < (2 * 4 * 8 * 512) / 8; i += blockDim.x)
            s[i] = g[i];
    }
#endif
    __syncthreads();

    const int lane = threadIdx.x & 31;
    const int wave = threadIdx.x >> 5;
    const int tile = blockIdx.x * 8 + wave;
    if (tile >= NTILES) return;  // wave-uniform: EXEC stays all-ones for WMMA

    const int ebase = tile * 16;
    const int row   = lane & 15;
    const int e     = ebase + row;

    const long long srcn = edge_index[e];
    const long long tgtn = edge_index[N_EDGES + e];
    const float     wgt  = edge_weight[e];

    // Gather bf16 A operands: 16 edges x 128 features, src and tgt halves.
    const unsigned short* srow = xq + (long long)srcn * HIDDEN;
    const unsigned short* trow = xq + (long long)tgtn * HIDDEN;
    v16bf a_src[4], a_tgt[4];
#pragma unroll
    for (int ks = 0; ks < 4; ++ks) {
        a_src[ks] = load_a(srow, ks * 32, lane);
        a_tgt[ks] = load_a(trow, ks * 32, lane);
    }

    // Per-ntile epilogue scalars (bias, rank-1 weight row, W2 column slice).
    float b1v[8], w1v[8], w2v[8];
#pragma unroll
    for (int nt = 0; nt < 8; ++nt) {
        const int n = nt * 16 + (lane & 15);
        b1v[nt] = b1[n];
        w1v[nt] = w1c[n];
        w2v[nt] = W2[n];
    }

    // Edge-weight values broadcast per C-row: m = r + 8*(lane>=16)
    float wv[8];
#pragma unroll
    for (int r = 0; r < 8; ++r)
        wv[r] = __shfl(wgt, r + ((lane >> 4) << 3), 32);

    const float b2v = b2[0];
    float sarr[8];
#pragma unroll
    for (int r = 0; r < 8; ++r) sarr[r] = 0.0f;

#pragma unroll
    for (int nt = 0; nt < 8; ++nt) {
        // Accumulator starts from WMMA inline-0 SRC2: the chain begins as soon
        // as the first B tile lands in LDS regs; bias terms are folded into the
        // epilogue so their loads never gate the matrix ops.
        v8f c = {};
#pragma unroll
        for (int ks = 0; ks < 4; ++ks)
            c = __builtin_amdgcn_wmma_f32_16x16x32_bf16(
                    false, a_src[ks], false, load_b(ldsW, 0, ks, nt, lane),
                    (short)0, c, false, false);
#pragma unroll
        for (int ks = 0; ks < 4; ++ks)
            c = __builtin_amdgcn_wmma_f32_16x16x32_bf16(
                    false, a_tgt[ks], false, load_b(ldsW, 1, ks, nt, lane),
                    (short)0, c, false, false);
        // Epilogue: + (b1[n] + w*W1c[n]), ReLU, fold into score partials.
#pragma unroll
        for (int r = 0; r < 8; ++r) {
            float h = fmaxf(c[r] + fmaf(wv[r], w1v[nt], b1v[nt]), 0.0f);
            sarr[r] = fmaf(h, w2v[nt], sarr[r]);
        }
    }

    // Reduce over the 16 N-lanes within each half (wave32 butterflies).
#pragma unroll
    for (int r = 0; r < 8; ++r) {
#pragma unroll
        for (int off = 1; off < 16; off <<= 1)
            sarr[r] += __shfl_xor(sarr[r], off, 32);
    }

    // Route score for this lane's own edge row (row = lane&15).
    const int rsel = row & 7;
    float v = sarr[0];
    v = (rsel == 1) ? sarr[1] : v;
    v = (rsel == 2) ? sarr[2] : v;
    v = (rsel == 3) ? sarr[3] : v;
    v = (rsel == 4) ? sarr[4] : v;
    v = (rsel == 5) ? sarr[5] : v;
    v = (rsel == 6) ? sarr[6] : v;
    v = (rsel == 7) ? sarr[7] : v;
    float vo = __shfl_xor(v, 16, 32);
    float score = (((row >> 3) ^ (lane >> 4)) ? vo : v) + b2v;
    float es = __expf(score);

    if (lane < 16) {
        exp_scores[e] = es;
        atomicAdd(&sum_exp[tgtn], es);
    }
}

// ---------------------------------------------------------------------------
// Kernel 4: out[tgt] += alpha * x[src]. One wave per edge, float4 per lane.
// ---------------------------------------------------------------------------
__global__ void __launch_bounds__(256)
scatter_kernel(const float* __restrict__ x,
               const long long* __restrict__ edge_index,
               const float* __restrict__ exp_scores,
               const float* __restrict__ sum_exp,
               float* __restrict__ out) {
    const int lane = threadIdx.x & 31;
    const int e = blockIdx.x * 8 + (threadIdx.x >> 5);
    if (e >= N_EDGES) return;
    const long long srcn = edge_index[e];
    const long long tgtn = edge_index[N_EDGES + e];
    const float alpha = exp_scores[e] / fmaxf(sum_exp[tgtn], 1e-12f);
    float4 vsrc = *(const float4*)(x + (long long)srcn * HIDDEN + lane * 4);
    float* o = out + (long long)tgtn * HIDDEN + lane * 4;
    atomicAdd(o + 0, alpha * vsrc.x);
    atomicAdd(o + 1, alpha * vsrc.y);
    atomicAdd(o + 2, alpha * vsrc.z);
    atomicAdd(o + 3, alpha * vsrc.w);
}

// ---------------------------------------------------------------------------
extern "C" void kernel_launch(void* const* d_in, const int* in_sizes, int n_in,
                              void* d_out, int out_size, void* d_ws, size_t ws_size,
                              hipStream_t stream) {
    const float*     x  = (const float*)d_in[0];
    const long long* ei = (const long long*)d_in[1];
    const float*     ew = (const float*)d_in[2];
    const float*     W1 = (const float*)d_in[3];
    const float*     b1 = (const float*)d_in[4];
    const float*     W2 = (const float*)d_in[5];
    const float*     b2 = (const float*)d_in[6];
    float* out = (float*)d_out;

    char* ws = (char*)d_ws;
    size_t off = 0;
    unsigned short* xq  = (unsigned short*)(ws + off); off += (size_t)N_NODES * HIDDEN * 2; // 25.6 MB
    unsigned short* w1b = (unsigned short*)(ws + off); off += 2 * 4 * 8 * 512 * 2;          // 64 KB
    float* expsc  = (float*)(ws + off); off += (size_t)N_EDGES * 4;                          // 2.4 MB
    float* sumexp = (float*)(ws + off); off += (size_t)N_NODES * 4;                          // 0.4 MB

    pack_w1_kernel<<<1, 256, 0, stream>>>(W1, w1b);

    const long long nquads = (long long)N_NODES * HIDDEN / 4;  // 3.2M
    convert_x_kernel<<<(int)((nquads + 255) / 256), 256, 0, stream>>>(x, xq, out, sumexp);

    edge_score_kernel<<<(NTILES + 7) / 8, 256, 0, stream>>>(
        xq, w1b, ei, ew, b1, W1 + 256 * HIDDEN, W2, b2, expsc, sumexp);

    scatter_kernel<<<(N_EDGES + 7) / 8, 256, 0, stream>>>(x, ei, expsc, sumexp, out);
}
